// EvoformerPermuter_85813446574744
// MI455X (gfx1250) — matrix operand order repaired and checked by reading, not compiled, not run on hardware
//
#include <hip/hip_runtime.h>

// ---------------------------------------------------------------------------
// EvoformerPermuter pipeline for MI455X (gfx1250, wave32, WMMA)
//   B=64, N=512, D=256, E=128, 20 Sinkhorn iterations
// K (67 MB) is computed once, stays L2-resident; Sinkhorn runs in
// diagonal-scaling (u,v) form so K is only re-read, never rewritten.
// ---------------------------------------------------------------------------

typedef __attribute__((ext_vector_type(16))) __bf16 v16bf;
typedef __attribute__((ext_vector_type(8)))  __bf16 v8bf;
typedef __attribute__((ext_vector_type(8)))  float  v8f;
typedef __attribute__((ext_vector_type(4)))  float  v4f;   // native vector for NT store

#define BB 64
#define NN 512
#define DD 256
#define EE 128
#define SINK_ITERS 20
#define CCH 8              // i-chunks per batch in the column pass

union V16U { v16bf v; v8bf h[2]; };

__device__ inline v16bf load_frag16(const __bf16* p0, const __bf16* p1) {
  V16U u;
  u.h[0] = *(const v8bf*)p0;
  u.h[1] = *(const v8bf*)p1;
  return u.v;
}

// --------------------------- prep kernels ----------------------------------

// x_in = where(mask, pad, emb_in); x_out = emb_out + pos  -> bf16
// emb_in / emb_out are single-use streams -> nontemporal loads (don't pollute L2)
__global__ void __launch_bounds__(256)
prep_inputs(const float* __restrict__ emb_in,
            const int*   __restrict__ mask,
            const float* __restrict__ emb_out,
            const float* __restrict__ pad,
            const float* __restrict__ pos,
            __bf16* __restrict__ xin,
            __bf16* __restrict__ xout) {
  size_t idx = (size_t)blockIdx.x * blockDim.x + threadIdx.x;
  const size_t total = (size_t)BB * NN * DD;
  const size_t stride = (size_t)gridDim.x * blockDim.x;
  for (; idx < total; idx += stride) {
    size_t bn = idx / DD;
    int d = (int)(idx % DD);
    int n = (int)(bn % NN);
    float ei = __builtin_nontemporal_load(&emb_in[idx]);
    float eo = __builtin_nontemporal_load(&emb_out[idx]);
    float xi = mask[bn] ? pad[d] : ei;
    float xo = eo + pos[(size_t)n * DD + d];
    xin[idx]  = (__bf16)xi;
    xout[idx] = (__bf16)xo;
  }
}

// W [D,E] fp32 -> Wt [E,D] bf16 (so WMMA B-operand rows are contiguous)
__global__ void __launch_bounds__(256)
prep_wt(const float* __restrict__ W, __bf16* __restrict__ Wt) {
  int idx = blockIdx.x * blockDim.x + threadIdx.x;
  if (idx < DD * EE) {
    int d = idx / EE, e = idx % EE;
    Wt[(size_t)e * DD + d] = (__bf16)W[idx];
  }
}

// --------------------------- WMMA GEMMs ------------------------------------

// Out[b,i,e] = sum_d X[b,i,d] * Wt[e,d]   (optionally * scale[e]), bf16 out
__global__ void __launch_bounds__(128)
gemm_xw(const __bf16* __restrict__ X,    // [B,N,D] bf16
        const __bf16* __restrict__ Wt,   // [E,D]   bf16
        const float*  __restrict__ scale,// [E] or nullptr
        __bf16* __restrict__ Out)        // [B,N,E] bf16
{
  const int lane = threadIdx.x & 31;
  const int wid  = threadIdx.x >> 5;
  const int b    = blockIdx.z;
  const int tile = blockIdx.x * 4 + wid;        // 256 tiles per batch
  const int tE   = EE / 16;                     // 8
  const int i0   = (tile / tE) * 16;
  const int e0   = (tile % tE) * 16;

  const int l16 = lane & 15;
  const int hiA = (lane < 16) ? 0 : 8;          // A 16x32 K-split
  const int hiB = (lane < 16) ? 0 : 8;          // B 32x16 K-split (same as A)
  const __bf16* arow = X  + ((size_t)b * NN + (size_t)(i0 + l16)) * DD;
  const __bf16* bcol = Wt + (size_t)(e0 + l16) * DD;

  v8f acc = {};
  #pragma unroll 4
  for (int k = 0; k < DD; k += 32) {
    __builtin_prefetch(arow + k + 64, 0, 1);
    v16bf af = load_frag16(arow + k + hiA, arow + k + 16 + hiA);
    v16bf bf = load_frag16(bcol + k + hiB, bcol + k + 16 + hiB);
    acc = __builtin_amdgcn_wmma_f32_16x16x32_bf16(false, af, false, bf,
                                                  (short)0, acc, false, false);
  }
  const int col = e0 + l16;
  const float s = scale ? scale[col] : 1.0f;
  #pragma unroll
  for (int r = 0; r < 8; ++r) {
    int row = i0 + r + ((lane < 16) ? 0 : 8);
    Out[((size_t)b * NN + row) * EE + col] = (__bf16)(acc[r] * s);
  }
}

// aff[b,i,j] = sum_e A[b,i,e]*Bm[b,j,e] + b_aff   (fp32 out)
__global__ void __launch_bounds__(128)
gemm_aff(const __bf16* __restrict__ A,   // [B,N,E] bf16
         const __bf16* __restrict__ Bm,  // [B,N,E] bf16
         const float*  __restrict__ baff,
         float* __restrict__ aff)        // [B,N,N] fp32
{
  const int lane = threadIdx.x & 31;
  const int wid  = threadIdx.x >> 5;
  const int b    = blockIdx.z;
  const int tile = blockIdx.x * 4 + wid;        // 1024 tiles per batch
  const int tJ   = NN / 16;                     // 32
  const int i0   = (tile / tJ) * 16;
  const int j0   = (tile % tJ) * 16;

  const int l16 = lane & 15;
  const int hiA = (lane < 16) ? 0 : 8;
  const int hiB = (lane < 16) ? 0 : 8;
  const __bf16* arow = A  + ((size_t)b * NN + (size_t)(i0 + l16)) * EE;
  const __bf16* brow = Bm + ((size_t)b * NN + (size_t)(j0 + l16)) * EE;

  v8f acc = {};
  #pragma unroll
  for (int k = 0; k < EE; k += 32) {
    v16bf af = load_frag16(arow + k + hiA, arow + k + 16 + hiA);
    v16bf bf = load_frag16(brow + k + hiB, brow + k + 16 + hiB);
    acc = __builtin_amdgcn_wmma_f32_16x16x32_bf16(false, af, false, bf,
                                                  (short)0, acc, false, false);
  }
  const float bias = baff[0];
  const int j = j0 + l16;
  #pragma unroll
  for (int r = 0; r < 8; ++r) {
    int i = i0 + r + ((lane < 16) ? 0 : 8);
    aff[((size_t)b * NN + i) * NN + j] = acc[r] + bias;
  }
}

// --------------------------- softmax stats ---------------------------------

// per (b,i): online max/sumexp over j (softmax axis=2). one wave per row, b128 loads.
__global__ void __launch_bounds__(256)
row_stats(const float* __restrict__ aff,
          float* __restrict__ rmax, float* __restrict__ rsum) {
  int lane = threadIdx.x & 31;
  int wid  = threadIdx.x >> 5;
  int row  = blockIdx.x * 8 + wid;              // b*N + i
  const float4* p = (const float4*)(aff + (size_t)row * NN);
  float m = -3.4e38f, s = 0.f;
  #pragma unroll
  for (int it = 0; it < NN / 128; ++it) {       // 4 iterations
    float4 kv = p[it * 32 + lane];
    float vals[4] = {kv.x, kv.y, kv.z, kv.w};
    #pragma unroll
    for (int q = 0; q < 4; ++q) {
      float nm = fmaxf(m, vals[q]);
      s = s * __expf(m - nm) + __expf(vals[q] - nm);
      m = nm;
    }
  }
  for (int off = 16; off > 0; off >>= 1) {
    float m2 = __shfl_xor(m, off);
    float s2 = __shfl_xor(s, off);
    float nm = fmaxf(m, m2);
    s = s * __expf(m - nm) + s2 * __expf(m2 - nm);
    m = nm;
  }
  if (lane == 0) { rmax[row] = m; rsum[row] = s; }
}

// per (b,j): online max/sumexp over i (softmax axis=1). coalesced across j.
__global__ void __launch_bounds__(256)
col_stats(const float* __restrict__ aff,
          float* __restrict__ cmax, float* __restrict__ csum) {
  int b = blockIdx.x >> 1;
  int j = ((blockIdx.x & 1) << 8) + threadIdx.x;
  const float* p = aff + (size_t)b * NN * NN + j;
  float m = -3.4e38f, s = 0.f;
  for (int i = 0; i < NN; ++i) {
    float v = p[(size_t)i * NN];
    float nm = fmaxf(m, v);
    s = s * __expf(m - nm) + __expf(v - nm);
    m = nm;
  }
  cmax[b * NN + j] = m;
  csum[b * NN + j] = s;
}

// K = 0.5*softmax_ax1 + 0.5*softmax_ax2  (in place over aff), float4 wide
__global__ void __launch_bounds__(256)
build_K(float* __restrict__ aff,
        const float* __restrict__ rmax, const float* __restrict__ rsum,
        const float* __restrict__ cmax, const float* __restrict__ csum) {
  size_t q = (size_t)blockIdx.x * blockDim.x + threadIdx.x;   // float4 index
  const size_t total4 = (size_t)BB * NN * NN / 4;
  const size_t stride = (size_t)gridDim.x * blockDim.x;
  float4* aff4 = (float4*)aff;
  for (; q < total4; q += stride) {
    size_t bi = q / (NN / 4);            // b*N + i
    int j = (int)(q % (NN / 4)) * 4;
    size_t b = bi / NN;
    float4 v  = aff4[q];
    float  rm = rmax[bi], rs = rsum[bi];
    float4 cm = *(const float4*)(cmax + b * NN + j);
    float4 cs = *(const float4*)(csum + b * NN + j);
    float4 o;
    o.x = 0.5f * __expf(v.x - rm) / rs + 0.5f * __expf(v.x - cm.x) / cs.x;
    o.y = 0.5f * __expf(v.y - rm) / rs + 0.5f * __expf(v.y - cm.y) / cs.y;
    o.z = 0.5f * __expf(v.z - rm) / rs + 0.5f * __expf(v.z - cm.z) / cs.z;
    o.w = 0.5f * __expf(v.w - rm) / rs + 0.5f * __expf(v.w - cm.w) / cs.w;
    aff4[q] = o;
  }
}

// --------------------------- Sinkhorn (scaling-vector form) ----------------

__global__ void __launch_bounds__(256)
init_ones(float* __restrict__ v, int n) {
  int i = blockIdx.x * blockDim.x + threadIdx.x;
  if (i < n) v[i] = 1.0f;
}

// u[b,i] = 1 / sum_j K[b,i,j]*v[b,j].  One wave per row, b128 loads,
// v[b,:] staged in LDS once per block (8 rows/block, all in same batch).
__global__ void __launch_bounds__(256)
row_mv(const float* __restrict__ K, const float* __restrict__ v,
       float* __restrict__ u) {
  __shared__ float vs[NN];
  const int tid = threadIdx.x;
  const int b   = blockIdx.x >> 6;              // 64 blocks per batch
  vs[tid]       = v[b * NN + tid];
  vs[tid + 256] = v[b * NN + tid + 256];
  __syncthreads();

  const int lane = tid & 31, wid = tid >> 5;
  const int row  = blockIdx.x * 8 + wid;        // b*N + i
  const float4* p = (const float4*)(K + (size_t)row * NN);
  float s = 0.f;
  #pragma unroll
  for (int it = 0; it < NN / 128; ++it) {       // 4 iterations
    int q = it * 32 + lane;
    float4 kv = p[q];
    int j = q * 4;
    s += kv.x * vs[j] + kv.y * vs[j + 1] + kv.z * vs[j + 2] + kv.w * vs[j + 3];
  }
  for (int off = 16; off > 0; off >>= 1) s += __shfl_xor(s, off);
  if (lane == 0) u[row] = 1.0f / s;
}

// partial[b,c,j] = sum_{i in chunk c} K[b,i,j]*u[b,i]; 512 concurrent blocks,
// u chunk staged in LDS, fully-coalesced b128 streaming of K columns.
__global__ void __launch_bounds__(128)
col_mv_partial(const float* __restrict__ K, const float* __restrict__ u,
               float* __restrict__ part) {
  __shared__ float us[NN / CCH];                // 64 floats
  const int b = blockIdx.x, c = blockIdx.y;
  const int tid = threadIdx.x;
  if (tid < NN / CCH) us[tid] = u[b * NN + c * (NN / CCH) + tid];
  __syncthreads();

  const int j0 = tid * 4;
  const float* base = K + (size_t)b * NN * NN + (size_t)(c * (NN / CCH)) * NN;
  float4 acc = {0.f, 0.f, 0.f, 0.f};
  #pragma unroll 4
  for (int i = 0; i < NN / CCH; ++i) {
    float4 kv = *(const float4*)(base + (size_t)i * NN + j0);
    float uu = us[i];
    acc.x += kv.x * uu; acc.y += kv.y * uu;
    acc.z += kv.z * uu; acc.w += kv.w * uu;
  }
  *(float4*)(part + ((size_t)b * CCH + c) * NN + j0) = acc;
}

// v[b,j] = 1 / sum_c partial[b,c,j]
__global__ void __launch_bounds__(256)
col_mv_reduce(const float* __restrict__ part, float* __restrict__ v) {
  int idx = blockIdx.x * blockDim.x + threadIdx.x;   // over B*N
  int b = idx >> 9, j = idx & (NN - 1);
  float s = 0.f;
  #pragma unroll
  for (int c = 0; c < CCH; ++c) s += part[((size_t)b * CCH + c) * NN + j];
  v[idx] = 1.0f / s;
}

// out[b,i,j] = u[b,i]*K[b,i,j]*v[b,j]; nontemporal b128 stores (write-once HBM)
__global__ void __launch_bounds__(256)
finalize(const float* __restrict__ K, const float* __restrict__ u,
         const float* __restrict__ v, float* __restrict__ out) {
  size_t q = (size_t)blockIdx.x * blockDim.x + threadIdx.x;   // float4 index
  const size_t total4 = (size_t)BB * NN * NN / 4;
  const size_t stride = (size_t)gridDim.x * blockDim.x;
  const v4f* K4 = (const v4f*)K;
  v4f* out4 = (v4f*)out;
  for (; q < total4; q += stride) {
    size_t bi = q / (NN / 4);
    int j = (int)(q % (NN / 4)) * 4;
    size_t b = bi / NN;
    v4f kv = K4[q];
    float ui = u[bi];
    v4f vj = *(const v4f*)(v + b * NN + j);
    v4f o = ui * kv * vj;                 // elementwise on native vectors
    __builtin_nontemporal_store(o, out4 + q);
  }
}

// --------------------------- host launch -----------------------------------

extern "C" void kernel_launch(void* const* d_in, const int* in_sizes, int n_in,
                              void* d_out, int out_size, void* d_ws, size_t ws_size,
                              hipStream_t stream) {
  const float* emb_in  = (const float*)d_in[0];
  const int*   mask    = (const int*)  d_in[1];
  const float* emb_out = (const float*)d_in[2];
  const float* pad     = (const float*)d_in[3];
  const float* pos     = (const float*)d_in[4];
  const float* Wa      = (const float*)d_in[5];
  const float* Wb      = (const float*)d_in[6];
  const float* waff    = (const float*)d_in[7];
  const float* baff    = (const float*)d_in[8];
  float* out = (float*)d_out;

  char* ws = (char*)d_ws;
  // aff/K (67,108,864 B) aliases the bf16 xin/xout staging (dead by then)
  float*  aff  = (float*) (ws + 0);
  __bf16* xin  = (__bf16*)(ws + 0);
  __bf16* xout = (__bf16*)(ws + 16777216);
  __bf16* Abf  = (__bf16*)(ws + 67108864);   //  8,388,608 B
  __bf16* Bbf  = (__bf16*)(ws + 75497472);   //  8,388,608 B
  __bf16* WaT  = (__bf16*)(ws + 83886080);   //     65,536 B
  __bf16* WbT  = (__bf16*)(ws + 83951616);   //     65,536 B
  float*  rmax = (float*) (ws + 84017152);
  float*  rsum = (float*) (ws + 84148224);
  float*  cmax = (float*) (ws + 84279296);
  float*  csum = (float*) (ws + 84410368);
  float*  uvec = (float*) (ws + 84541440);
  float*  vvec = (float*) (ws + 84672512);
  float*  part = (float*) (ws + 84803584);   //  1,048,576 B (col-pass partials)

  // 1) prep: masked inputs / pos-encoded outputs -> bf16, W transposes -> bf16
  prep_inputs<<<8192, 256, 0, stream>>>(emb_in, mask, emb_out, pad, pos, xin, xout);
  prep_wt<<<(DD * EE + 255) / 256, 256, 0, stream>>>(Wa, WaT);
  prep_wt<<<(DD * EE + 255) / 256, 256, 0, stream>>>(Wb, WbT);

  // 2) A = (x_out @ Wa) * w_aff ; Bm = x_in @ Wb   (WMMA bf16)
  dim3 g1(64, 1, BB);   // 256 tiles/batch, 4 waves/block
  gemm_xw<<<g1, 128, 0, stream>>>(xout, WaT, waff, Abf);
  gemm_xw<<<g1, 128, 0, stream>>>(xin,  WbT, nullptr, Bbf);

  // 3) aff = A @ Bm^T + b_aff   (WMMA bf16, fp32 out)
  dim3 g2(256, 1, BB);  // 1024 tiles/batch
  gemm_aff<<<g2, 128, 0, stream>>>(Abf, Bbf, baff, aff);

  // 4) dual softmax mix -> K (in place)
  row_stats<<<BB * NN / 8, 256, 0, stream>>>(aff, rmax, rsum);
  col_stats<<<BB * 2, 256, 0, stream>>>(aff, cmax, csum);
  build_K<<<8192, 256, 0, stream>>>(aff, rmax, rsum, cmax, csum);

  // 5) Sinkhorn, diagonal-scaling form: K stays L2-resident, never rewritten
  init_ones<<<(BB * NN + 255) / 256, 256, 0, stream>>>(vvec, BB * NN);
  dim3 gcp(BB, CCH);
  for (int t = 0; t < SINK_ITERS; ++t) {
    row_mv<<<BB * NN / 8, 256, 0, stream>>>(aff, vvec, uvec);
    col_mv_partial<<<gcp, 128, 0, stream>>>(aff, uvec, part);
    col_mv_reduce<<<BB * NN / 256, 256, 0, stream>>>(part, vvec);
  }

  // 6) P = diag(u) K diag(v)
  finalize<<<8192, 256, 0, stream>>>(aff, uvec, vvec, out);
}
